// RNN_60919816127104
// MI455X (gfx1250) — matrix operand order repaired and checked by reading, not compiled, not run on hardware
//
#include <hip/hip_runtime.h>
#include <hip/hip_bf16.h>
#include <stdint.h>

#define BDIM 8
#define TDIM 1024
#define EDIM 1024
#define HDIM 1024
#define VDIM 8192

typedef __attribute__((ext_vector_type(16))) __bf16 v16bf;
typedef __attribute__((ext_vector_type(8)))  float  v8f;
typedef __attribute__((ext_vector_type(4)))  unsigned int v4u;
typedef __attribute__((ext_vector_type(8)))  int v8i;
typedef __attribute__((ext_vector_type(4)))  int v4i;

union FragU { uint4 u[2]; v16bf v; };

#if defined(__gfx1250__) && __has_builtin(__builtin_amdgcn_tensor_load_to_lds)
#define HAVE_TDM 1
#else
#define HAVE_TDM 0
#endif

// LDS tile geometry: 128 rows x 32 bf16, padded to 40 bf16 (80B) row stride.
// 80B stride keeps 16B alignment for b128 LDS loads and spreads banks.
#define TILE_ROWS 128
#define TILE_K    32
#define LDS_LDA   40

#if HAVE_TDM
#define LDS_OFF(p) ((unsigned)(uintptr_t)(__attribute__((address_space(3))) __bf16*)(p))

// Build a D# (CDNA5 ISA) for a 2D tile load:
//   tile = TILE_K x TILE_ROWS bf16 elements, tensor row stride = k_stride elems,
//   LDS padding: 4 DWORDs (16B) every 16 DWORDs (64B) -> 80B LDS row stride.
__device__ __forceinline__ void tdm_load_tile(unsigned lds_off,
                                              const __bf16* gsrc,
                                              unsigned k_stride) {
  unsigned long long ga = (unsigned long long)(uintptr_t)gsrc;
  const unsigned dim0 = k_stride;       // tensor_dim0 (elements) — OOB bound
  const unsigned dim1 = 1u << 20;       // tensor_dim1 — generous OOB bound
  v4u g0 = { 1u,                                    // count=1, user descriptor
             lds_off,                               // lds_addr
             (unsigned)ga,                          // global_addr[31:0]
             ((unsigned)(ga >> 32) & 0x01FFFFFFu)   // global_addr[56:32]
             | 0x80000000u };                       // type=2 ("image")
  const unsigned w0 = (1u << 16)   // data_size = 2 bytes
                    | (1u << 20)   // pad_enable
                    | (3u << 22)   // pad_interval: 16 DWORDs (64B)
                    | (3u << 25);  // pad_amount:   4 DWORDs (16B)
  v8i g1 = { (int)w0,
             (int)((dim0 & 0xFFFFu) << 16),                     // dim0[15:0]
             (int)((dim0 >> 16) | ((dim1 & 0xFFFFu) << 16)),    // dim0[31:16], dim1[15:0]
             (int)((dim1 >> 16) | ((unsigned)TILE_K << 16)),    // dim1[31:16], tile_dim0
             (int)TILE_ROWS,                                    // tile_dim1; tile_dim2=0
             (int)k_stride,                                     // tensor_dim0_stride lo32
             0, 0 };                                            // stride hi, dim1_stride=0
  v4i zz4 = { 0, 0, 0, 0 };
  v8i zz8 = { 0, 0, 0, 0, 0, 0, 0, 0 };
  // clang-23 / therock-10.0 signature: (u32x4, i32x8, i32x4, i32x4, i32x8, i32 cpol)
  __builtin_amdgcn_tensor_load_to_lds(g0, g1, zz4, zz4, zz8, 0);
}
#endif

// Synchronous fallback staging (also used if TDM builtin is unavailable)
__device__ __forceinline__ void stage_tile(const __bf16* __restrict__ gbase,
                                           int k_stride,
                                           __bf16* __restrict__ lds, int tid) {
  // 128 rows x 4 chunks of 16B per row
  for (int c = tid; c < TILE_ROWS * 4; c += 256) {
    const int row = c >> 2, ch = c & 3;
    uint4 v = *(const uint4*)(gbase + (size_t)row * k_stride + ch * 8);
    *(uint4*)(lds + row * LDS_LDA + ch * 8) = v;
  }
}

// ---------------------------------------------------------------------------
// f32 -> bf16 (round to nearest even), grid-strided
// ---------------------------------------------------------------------------
__global__ void f32_to_bf16(const float* __restrict__ in,
                            __bf16* __restrict__ out, long long n) {
  long long i = (long long)blockIdx.x * blockDim.x + threadIdx.x;
  long long stride = (long long)gridDim.x * blockDim.x;
  for (; i < n; i += stride) {
    unsigned u = __float_as_uint(in[i]);
    unsigned rnd = 0x7FFFu + ((u >> 16) & 1u);
    unsigned short h = (unsigned short)((u + rnd) >> 16);
    out[i] = __builtin_bit_cast(__bf16, h);
  }
}

// ---------------------------------------------------------------------------
// Zero the ping-pong h-state + barrier counter (graph-replay determinism)
// ---------------------------------------------------------------------------
__global__ void init_state(__bf16* __restrict__ hstate,
                           unsigned* __restrict__ bar) {
  int i = blockIdx.x * blockDim.x + threadIdx.x;
  if (i == 0) *bar = 0u;
  const int n = 2 * 16 * HDIM;
  for (; i < n; i += gridDim.x * blockDim.x)
    hstate[i] = __builtin_bit_cast(__bf16, (unsigned short)0);
}

// ---------------------------------------------------------------------------
// TDM-staged WMMA GEMM:  Out[M,N] = A[M,K] * Bw[N,K]^T + bias
// Block = 256 threads (8 waves), tile 128x128, K-step 32, double-buffered LDS
// fed by tensor_load_to_lds (TENSORcnt + barrier pipelining).
// ---------------------------------------------------------------------------
__global__ __launch_bounds__(256)
void wmma_gemm_staged(const __bf16* __restrict__ A,
                      const __bf16* __restrict__ Bw,
                      const float* __restrict__ bias,
                      float* __restrict__ Out,
                      int M, int N, int K) {
  __shared__ __bf16 shA[2][TILE_ROWS * LDS_LDA];
  __shared__ __bf16 shB[2][TILE_ROWS * LDS_LDA];

  const int lane = threadIdx.x & 31;
  const int wave = threadIdx.x >> 5;  // 0..7
  const int wm = wave >> 1;           // 0..3
  const int wn = wave & 1;            // 0..1
  const int g  = lane >> 4;
  const int lr = lane & 15;
  const int m_blk = blockIdx.y * 128;
  const int n_blk = blockIdx.x * 128;
  const int nk = K / TILE_K;

  v8f acc[2][4];
#pragma unroll
  for (int i = 0; i < 2; ++i)
#pragma unroll
    for (int j = 0; j < 4; ++j) acc[i][j] = (v8f){};

#if HAVE_TDM
  if (threadIdx.x < 32) {
    tdm_load_tile(LDS_OFF(&shA[0][0]), A + (size_t)m_blk * K, (unsigned)K);
    tdm_load_tile(LDS_OFF(&shB[0][0]), Bw + (size_t)n_blk * K, (unsigned)K);
  }
#endif

  for (int kt = 0; kt < nk; ++kt) {
    const int buf = kt & 1;
#if HAVE_TDM
    if (threadIdx.x < 32) __builtin_amdgcn_s_wait_tensorcnt(0);
    __syncthreads();  // buf's tiles visible to all waves
    if (kt + 1 < nk && threadIdx.x < 32) {
      const int k1 = (kt + 1) * TILE_K;
      tdm_load_tile(LDS_OFF(&shA[1 - buf][0]), A + (size_t)m_blk * K + k1, (unsigned)K);
      tdm_load_tile(LDS_OFF(&shB[1 - buf][0]), Bw + (size_t)n_blk * K + k1, (unsigned)K);
    }
#else
    __syncthreads();
    stage_tile(A + (size_t)m_blk * K + kt * TILE_K, K, &shA[buf][0], threadIdx.x);
    stage_tile(Bw + (size_t)n_blk * K + kt * TILE_K, K, &shB[buf][0], threadIdx.x);
    __syncthreads();
#endif

    v16bf a[2], b[4];
#pragma unroll
    for (int i = 0; i < 2; ++i) {
      const __bf16* p = &shA[buf][(wm * 32 + i * 16 + lr) * LDS_LDA];
      FragU f;
      f.u[0] = *(const uint4*)(p + g * 8);
      f.u[1] = *(const uint4*)(p + 16 + g * 8);
      a[i] = f.v;
    }
#pragma unroll
    for (int j = 0; j < 4; ++j) {
      const __bf16* p = &shB[buf][(wn * 64 + j * 16 + lr) * LDS_LDA];
      FragU f;
      f.u[0] = *(const uint4*)(p + g * 8);
      f.u[1] = *(const uint4*)(p + 16 + g * 8);
      b[j] = f.v;
    }
#pragma unroll
    for (int i = 0; i < 2; ++i)
#pragma unroll
      for (int j = 0; j < 4; ++j)
        acc[i][j] = __builtin_amdgcn_wmma_f32_16x16x32_bf16(
            false, a[i], false, b[j], (short)0, acc[i][j], false, false);

    __syncthreads();  // all waves done reading buf before it is refilled
  }

#pragma unroll
  for (int j = 0; j < 4; ++j) {
    const int col = n_blk + wn * 64 + j * 16 + lr;
    const float bv = bias[col];
#pragma unroll
    for (int i = 0; i < 2; ++i) {
#pragma unroll
      for (int r = 0; r < 8; ++r) {
        const int row = m_blk + wm * 32 + i * 16 + r + g * 8;
        Out[(size_t)row * N + col] = acc[i][j][r] + bv;
      }
    }
  }
}

// ---------------------------------------------------------------------------
// Direct-global WMMA GEMM with row gather (embedding rows via X indices):
// Out[M,N] = A[idx[m],K] * Bw[N,K]^T + bias. Same wave tiling as above.
// ---------------------------------------------------------------------------
__global__ __launch_bounds__(256)
void wmma_gemm_gather(const __bf16* __restrict__ A,
                      const int* __restrict__ rowidx,
                      const __bf16* __restrict__ Bw,
                      const float* __restrict__ bias,
                      float* __restrict__ Out,
                      int M, int N, int K) {
  const int lane = threadIdx.x & 31;
  const int wave = threadIdx.x >> 5;
  const int wm = wave >> 1;
  const int wn = wave & 1;
  const int g  = lane >> 4;
  const int lr = lane & 15;
  const int m_base = blockIdx.y * 128 + wm * 32;
  const int n_base = blockIdx.x * 128 + wn * 64;

  const __bf16* arow[2];
#pragma unroll
  for (int i = 0; i < 2; ++i)
    arow[i] = A + (size_t)rowidx[m_base + i * 16 + lr] * K;
  const __bf16* brow[4];
#pragma unroll
  for (int j = 0; j < 4; ++j)
    brow[j] = Bw + (size_t)(n_base + j * 16 + lr) * K;

  v8f acc[2][4];
#pragma unroll
  for (int i = 0; i < 2; ++i)
#pragma unroll
    for (int j = 0; j < 4; ++j) acc[i][j] = (v8f){};

#pragma unroll 2
  for (int k0 = 0; k0 < K; k0 += 32) {
    v16bf a[2], b[4];
#pragma unroll
    for (int i = 0; i < 2; ++i) {
      FragU f;
      f.u[0] = *(const uint4*)(arow[i] + k0 + g * 8);
      f.u[1] = *(const uint4*)(arow[i] + k0 + 16 + g * 8);
      a[i] = f.v;
    }
#pragma unroll
    for (int j = 0; j < 4; ++j) {
      FragU f;
      f.u[0] = *(const uint4*)(brow[j] + k0 + g * 8);
      f.u[1] = *(const uint4*)(brow[j] + k0 + 16 + g * 8);
      b[j] = f.v;
    }
#pragma unroll
    for (int i = 0; i < 2; ++i)
#pragma unroll
      for (int j = 0; j < 4; ++j)
        acc[i][j] = __builtin_amdgcn_wmma_f32_16x16x32_bf16(
            false, a[i], false, b[j], (short)0, acc[i][j], false, false);
  }

#pragma unroll
  for (int j = 0; j < 4; ++j) {
    const int col = n_base + j * 16 + lr;
    const float bv = bias[col];
#pragma unroll
    for (int i = 0; i < 2; ++i) {
#pragma unroll
      for (int r = 0; r < 8; ++r) {
        const int row = m_base + i * 16 + r + g * 8;
        Out[(size_t)row * N + col] = acc[i][j][r] + bv;
      }
    }
  }
}

// ---------------------------------------------------------------------------
// Linear recurrence: h_t = xh_t + h_{t-1} @ Whh^T + bhh (batch padded 8->16).
// 16 persistent WGs x 128 threads; WG w owns 64 columns; one 16x16 WMMA tile
// per wave, K-chained over 32 WMMAs; ping-pong h-state; atomic grid barrier.
// ---------------------------------------------------------------------------
__global__ __launch_bounds__(128)
void rnn_recurrence(const float* __restrict__ xh,
                    const __bf16* __restrict__ Whh,
                    const float* __restrict__ bhh,
                    __bf16* __restrict__ hstate,  // 2 buffers of 16*HDIM bf16
                    __bf16* __restrict__ hs,      // [B,T,H] bf16
                    unsigned* __restrict__ bar) {
  const int lane = threadIdx.x & 31;
  const int wave = threadIdx.x >> 5;  // 0..3
  const int g  = lane >> 4;
  const int lr = lane & 15;
  const int ncol = blockIdx.x * 64 + wave * 16 + lr;

  const __bf16* brow = Whh + (size_t)ncol * HDIM;
  const float bv = bhh[ncol];

  for (int t = 0; t < TDIM; ++t) {
    const __bf16* hr = hstate + (size_t)(t & 1) * (16 * HDIM);
    __bf16* hw       = hstate + (size_t)((t + 1) & 1) * (16 * HDIM);
    const __bf16* arow = hr + (size_t)lr * HDIM;

    v8f acc = (v8f){};
#pragma unroll 8
    for (int k0 = 0; k0 < HDIM; k0 += 32) {
      FragU fa, fb;
      fa.u[0] = *(const uint4*)(arow + k0 + g * 8);
      fa.u[1] = *(const uint4*)(arow + k0 + 16 + g * 8);
      fb.u[0] = *(const uint4*)(brow + k0 + g * 8);
      fb.u[1] = *(const uint4*)(brow + k0 + 16 + g * 8);
      acc = __builtin_amdgcn_wmma_f32_16x16x32_bf16(
          false, fa.v, false, fb.v, (short)0, acc, false, false);
    }

#pragma unroll
    for (int r = 0; r < 8; ++r) {
      const int row = r + g * 8;
      if (row < BDIM) {
        float v = acc[r] + xh[((size_t)row * TDIM + t) * HDIM + ncol] + bv;
        unsigned u = __float_as_uint(v);
        unsigned rnd = 0x7FFFu + ((u >> 16) & 1u);
        unsigned short hb = (unsigned short)((u + rnd) >> 16);
        __bf16 hv = __builtin_bit_cast(__bf16, hb);
        hw[(size_t)row * HDIM + ncol] = hv;
        hs[((size_t)row * TDIM + t) * HDIM + ncol] = hv;
      } else {
        hw[(size_t)row * HDIM + ncol] = __builtin_bit_cast(__bf16, (unsigned short)0);
      }
    }

    __threadfence();
    __syncthreads();
    if (threadIdx.x == 0) {
      __hip_atomic_fetch_add(bar, 1u, __ATOMIC_RELEASE, __HIP_MEMORY_SCOPE_AGENT);
      const unsigned target = 16u * (unsigned)(t + 1);
      while (__hip_atomic_load(bar, __ATOMIC_ACQUIRE, __HIP_MEMORY_SCOPE_AGENT) < target) {
        __builtin_amdgcn_s_sleep(2);
      }
    }
    __syncthreads();
  }
}

// ---------------------------------------------------------------------------
// Host driver
// ---------------------------------------------------------------------------
extern "C" void kernel_launch(void* const* d_in, const int* in_sizes, int n_in,
                              void* d_out, int out_size, void* d_ws, size_t ws_size,
                              hipStream_t stream) {
  (void)in_sizes; (void)n_in; (void)out_size; (void)ws_size;
  const int*   X   = (const int*)d_in[0];
  const float* Emb = (const float*)d_in[1];
  const float* Wxh = (const float*)d_in[2];
  const float* bxh = (const float*)d_in[3];
  const float* Whh = (const float*)d_in[4];
  const float* bhh = (const float*)d_in[5];
  const float* Wyh = (const float*)d_in[6];
  const float* byh = (const float*)d_in[7];
  float* out = (float*)d_out;

  char* ws = (char*)d_ws;
  size_t off = 0;
  __bf16* EmbB = (__bf16*)(ws + off); off += (size_t)VDIM * EDIM * 2;
  __bf16* WxhB = (__bf16*)(ws + off); off += (size_t)HDIM * EDIM * 2;
  __bf16* WhhB = (__bf16*)(ws + off); off += (size_t)HDIM * HDIM * 2;
  __bf16* WyhB = (__bf16*)(ws + off); off += (size_t)VDIM * HDIM * 2;
  float*  xh   = (float*)(ws + off);  off += (size_t)BDIM * TDIM * HDIM * 4;
  __bf16* hs   = (__bf16*)(ws + off); off += (size_t)BDIM * TDIM * HDIM * 2;
  __bf16* hst  = (__bf16*)(ws + off); off += (size_t)2 * 16 * HDIM * 2;
  unsigned* bar = (unsigned*)(ws + off); off += 256;

  f32_to_bf16<<<1024, 256, 0, stream>>>(Emb, EmbB, (long long)VDIM * EDIM);
  f32_to_bf16<<<256,  256, 0, stream>>>(Wxh, WxhB, (long long)HDIM * EDIM);
  f32_to_bf16<<<256,  256, 0, stream>>>(Whh, WhhB, (long long)HDIM * HDIM);
  f32_to_bf16<<<1024, 256, 0, stream>>>(Wyh, WyhB, (long long)VDIM * HDIM);
  init_state<<<64, 256, 0, stream>>>(hst, bar);

  // xh = Emb[X] @ Wxh^T + bxh (gathered GEMM)
  wmma_gemm_gather<<<dim3(HDIM / 128, (BDIM * TDIM) / 128), 256, 0, stream>>>(
      EmbB, X, WxhB, bxh, xh, BDIM * TDIM, HDIM, EDIM);

  // serialized linear recurrence
  rnn_recurrence<<<16, 128, 0, stream>>>(xh, WhhB, bhh, hst, hs, bar);

  // logits = hs @ Wyh^T + byh (TDM-staged GEMM, dominant cost)
  wmma_gemm_staged<<<dim3(VDIM / 128, (BDIM * TDIM) / 128), 256, 0, stream>>>(
      hs, WyhB, byh, out, BDIM * TDIM, VDIM, HDIM);
}
